// MaskedCrossAttention_59304908423585
// MI455X (gfx1250) — matrix operand order repaired and checked by reading, not compiled, not run on hardware
//
#include <hip/hip_runtime.h>
#include <hip/hip_bf16.h>

// ---------------------------------------------------------------------------
// MaskedCrossAttention for MI455X (gfx1250, wave32, WMMA bf16 16x16x32, TDM)
// ---------------------------------------------------------------------------

typedef __bf16 bf16;
typedef __attribute__((ext_vector_type(16))) __bf16        v16bf;
typedef __attribute__((ext_vector_type(8)))  float         v8f;
typedef __attribute__((ext_vector_type(4))) unsigned int   u32x4;
typedef __attribute__((ext_vector_type(8)))  int           i32x8;
typedef __attribute__((ext_vector_type(4)))  int           i32x4;
typedef __attribute__((address_space(3)))    bf16          lds_bf16;

#define B_SZ    4
#define T_TXT   2048
#define D_MODEL 4096
#define T_IMG   8
#define N_IMG   64
#define T_KEY   512      // T_IMG * N_IMG
#define DV      1024
#define HEADS   16
#define DHEAD   64
#define INNER   1024     // HEADS*DHEAD
#define SCALE_Q 0.125f   // 64^-0.5
#define INTMAXR 10000
#define NEG_BIG (-3.0e38f)
#define KPAD    72       // 64 + 8 bf16 pad per row (bank-conflict-free)

union FragU { u32x4 q[2]; v16bf v; };

// A fragment (16x32 bf16, M x K): lane m(=lane&15) holds row m,
// K = half*8 + {0..7}  and  16 + half*8 + {0..7}   (half = lane>>4)
static __device__ __forceinline__ v16bf load_frag_a(const bf16* p0, int lda, int lane) {
  const int half = lane >> 4, m = lane & 15;
  const bf16* p = p0 + (size_t)m * lda + half * 8;
  FragU f;
  f.q[0] = *(const u32x4*)(p);
  f.q[1] = *(const u32x4*)(p + 16);
  return f.v;
}

// B fragment (32x16 bf16, K x N) from N-major storage (row n holds K contiguous):
// lane n(=lane&15) holds column n, K = half*8 + {0..7} and 16 + half*8 + {0..7}
static __device__ __forceinline__ v16bf load_frag_b(const bf16* p0, int ldb, int lane) {
  const int half = lane >> 4, n = lane & 15;
  const bf16* p = p0 + (size_t)n * ldb + half * 8;
  FragU f;
  f.q[0] = *(const u32x4*)(p);
  f.q[1] = *(const u32x4*)(p + 16);
  return f.v;
}

static __device__ __forceinline__ v8f wmma_bf16(v16bf a, v16bf b, v8f c) {
  return __builtin_amdgcn_wmma_f32_16x16x32_bf16(false, a, false, b, (short)0, c,
                                                 false, false);
}

// ---------------------------------------------------------------------------
// LayerNorm: f32 (rows of 4096) -> bf16
// ---------------------------------------------------------------------------
__global__ __launch_bounds__(256) void ln_kernel(const float* __restrict__ x,
                                                 const float* __restrict__ g,
                                                 const float* __restrict__ be,
                                                 bf16* __restrict__ xn) {
  __shared__ float sh[256], sh2[256];
  const int row = blockIdx.x, tid = threadIdx.x;
  const float* xr = x + (size_t)row * D_MODEL;
  float s = 0.f, s2 = 0.f;
  for (int c = tid; c < D_MODEL; c += 256) { float v = xr[c]; s += v; s2 += v * v; }
  sh[tid] = s; sh2[tid] = s2;
  __syncthreads();
  for (int o = 128; o > 0; o >>= 1) {
    if (tid < o) { sh[tid] += sh[tid + o]; sh2[tid] += sh2[tid + o]; }
    __syncthreads();
  }
  const float mean = sh[0] * (1.f / D_MODEL);
  const float var  = sh2[0] * (1.f / D_MODEL) - mean * mean;
  const float rstd = rsqrtf(var + 1e-5f);
  for (int c = tid; c < D_MODEL; c += 256)
    xn[(size_t)row * D_MODEL + c] = (bf16)((xr[c] - mean) * rstd * g[c] + be[c]);
}

// ---------------------------------------------------------------------------
// f32 -> bf16 convert / N-major transpose-convert of weights
// ---------------------------------------------------------------------------
__global__ void convert_bf16(const float* __restrict__ src, bf16* __restrict__ dst, int n) {
  int t = blockIdx.x * 256 + threadIdx.x;
  if (t < n) dst[t] = (bf16)src[t];
}

// W: (K,N) row-major f32  ->  WT: (N,K) row-major bf16
__global__ void transpose_bf16(const float* __restrict__ W, bf16* __restrict__ WT,
                               int K, int N) {
  int t = blockIdx.x * 256 + threadIdx.x;
  if (t < N * K) {
    int n = t / K, k = t % K;
    WT[t] = (bf16)W[(size_t)k * N + n];
  }
}

// ---------------------------------------------------------------------------
// chunk2mask: sequential per batch row (tiny: 4 rows x ~3 passes of 2048)
// ---------------------------------------------------------------------------
__global__ void mask_kernel(const unsigned char* __restrict__ ml,
                            const unsigned char* __restrict__ cl,
                            int* __restrict__ cm, int* __restrict__ im) {
  __shared__ int chs[B_SZ][T_TXT];
  const int b = threadIdx.x;
  if (b >= B_SZ) return;
  int* c = chs[b];
  int tt = 0;
  for (int t = 0; t < T_TXT; ++t) {
    tt += ml[b * T_TXT + t] ? 1 : 0;
    c[t] = cl[b * T_TXT + t] ? tt : 0;       // chunk_tensor row
  }
  const bool z0 = (c[0] == 0);
  int nxt = INTMAXR;                          // backward fill with next nonzero
  for (int t = T_TXT - 1; t >= 0; --t) {
    if (c[t] != 0) nxt = c[t];
    else c[t] = nxt;
  }
  if (z0) c[0] = 0;                           // chunk[0] special case
  for (int t = 0; t < T_TXT; ++t) cm[b * T_TXT + t] = c[t];
  // inverse mask: previous distinct value (searchsorted-left semantics)
  int first = c[0], cur = first, pd = first;
  bool firstRun = true;
  for (int t = 0; t < T_TXT; ++t) {
    int v = c[t];
    if (v != cur) { pd = cur; cur = v; firstRun = false; }
    int prev = firstRun ? first : pd;
    int iv = (v == 0) ? 0 : prev;
    if (t == 0) iv = INTMAXR;
    im[b * T_TXT + t] = iv;
  }
}

// ---------------------------------------------------------------------------
// Generic bf16 WMMA GEMM: A (M,K) row-major, Bn (N,K) N-major, f32 accumulate.
// Wave tile 64x64 (16 wmma per 8 fragment loads per 32-deep k-step);
// block = 8 waves (4x2) -> 256x128 tile.  Grid: (N/128, M/256)
// ---------------------------------------------------------------------------
template <typename Epi>
__global__ __launch_bounds__(256) void gemm_bf16_kernel(const bf16* __restrict__ A,
                                                        const bf16* __restrict__ Bn,
                                                        int K, Epi epi) {
  const int lane = threadIdx.x & 31;
  const int wave = threadIdx.x >> 5;
  const int m0 = blockIdx.y * 256 + (wave & 3) * 64;
  const int n0 = blockIdx.x * 128 + (wave >> 2) * 64;
  const bf16* ap[4];
#pragma unroll
  for (int mi = 0; mi < 4; ++mi) ap[mi] = A + (size_t)(m0 + 16 * mi) * K;
  v8f acc[4][4] = {};
  for (int k0 = 0; k0 < K; k0 += 32) {
    v16bf af[4];
#pragma unroll
    for (int mi = 0; mi < 4; ++mi) {
      af[mi] = load_frag_a(ap[mi] + k0, K, lane);
      __builtin_prefetch(ap[mi] + k0 + 128, 0, 0);   // global_prefetch_b8
    }
#pragma unroll
    for (int c = 0; c < 4; ++c) {
      v16bf bb = load_frag_b(Bn + (size_t)(n0 + 16 * c) * K + k0, K, lane);
#pragma unroll
      for (int mi = 0; mi < 4; ++mi)
        acc[mi][c] = wmma_bf16(af[mi], bb, acc[mi][c]);
    }
  }
  // C/D layout: elem r of acc -> (M = r + 8*(lane>=16), N = lane&15)
  const int half = lane >> 4, nl = lane & 15;
#pragma unroll
  for (int mi = 0; mi < 4; ++mi)
#pragma unroll
    for (int c = 0; c < 4; ++c)
#pragma unroll
      for (int r = 0; r < 8; ++r)
        epi(m0 + mi * 16 + r + 8 * half, n0 + 16 * c + nl, acc[mi][c][r]);
}

// Epilogues -----------------------------------------------------------------
struct EpiQ {   // q: scale and scatter to qh[b][h][i][d] bf16
  bf16* qh;
  __device__ void operator()(int m, int n, float v) const {
    int b = m >> 11, i = m & 2047, h = n >> 6, d = n & 63;
    qh[(((size_t)(b * HEADS + h) * T_TXT + i) << 6) + d] = (bf16)(v * SCALE_Q);
  }
};
struct EpiKV {  // k -> kh[b][h][j][d] ; v -> vT[b][h][d][j]
  bf16 *kh, *vt;
  __device__ void operator()(int m, int n, float v) const {
    int b = m >> 9, j = m & 511;
    if (n < INNER) {
      int h = n >> 6, d = n & 63;
      kh[(((size_t)(b * HEADS + h) * T_KEY + j) << 6) + d] = (bf16)v;
    } else {
      int n2 = n - INNER, h = n2 >> 6, d = n2 & 63;
      vt[((size_t)(b * HEADS + h) * DHEAD + d) * T_KEY + j] = (bf16)v;
    }
  }
};
struct EpiO {   // final projection -> f32 output
  float* out;
  __device__ void operator()(int m, int n, float v) const {
    out[(size_t)m * D_MODEL + n] = v;
  }
};

// ---------------------------------------------------------------------------
// Flash attention: wave owns 16 query rows x 512 keys, online softmax over
// 64-key chunks.  One TDM tensor_load_to_lds stages the whole K matrix of
// this block's (b,h) into LDS with hardware row padding (64 -> 72 bf16) so
// all K-fragment reads are bank-conflict-free.  QK^T and P*V via WMMA.
// Each block covers 8 row-tiles of one (b,h): 16 blocks per head.
// Grid: (B*HEADS*(T_TXT/16))/8 blocks of 256 threads.
// ---------------------------------------------------------------------------
__global__ __launch_bounds__(256) void attn_kernel(const bf16* __restrict__ qh,
                                                   const bf16* __restrict__ kh,
                                                   const bf16* __restrict__ vt,
                                                   const int*  __restrict__ cmv,
                                                   const int*  __restrict__ imv,
                                                   bf16* __restrict__ ob) {
  __shared__ __align__(16) bf16 Ksh[T_KEY * KPAD];            // 72 KB, TDM-padded
  __shared__ __align__(16) bf16 Pl[8][16][KPAD];              // 18 KB prob tiles
  const int lane = threadIdx.x & 31;
  const int wave = threadIdx.x >> 5;
  const int wid  = blockIdx.x * 8 + wave;
  const int b   = wid >> 11;            // 16 heads * 128 tiles per batch
  const int rem = wid & 2047;
  const int h   = rem >> 7;
  const int i0  = (rem & 127) << 4;
  const int half = lane >> 4, nl = lane & 15;
  bf16 (*P)[KPAD] = Pl[wave];

  const bf16* kbase = kh + ((size_t)(b * HEADS + h) << 15);   // *512*64
  const bf16* vbase = vt + ((size_t)(b * HEADS + h) << 15);

  // ---- TDM: DMA K[b][h] (512x64 bf16) into LDS, pad 8 bf16 per 64-elem row.
  // D# per CDNA5 ISA sec.8:
  //   group0: count=1 | lds_addr | global_addr(57b) | type=2
  //   group1: data_size=2B, pad_enable, pad_interval=32dw(code 4),
  //           pad_amount=4dw(code 3), tensor 64x512, tile 64x512, stride0=64
  if (wave == 0) {
    const unsigned long long ga = (unsigned long long)(uintptr_t)kbase;
    const unsigned ldsoff = (unsigned)(uintptr_t)(lds_bf16*)Ksh;
    u32x4 g0;
    g0[0] = 1u;                                        // count=1 (valid)
    g0[1] = ldsoff;                                    // lds_addr
    g0[2] = (unsigned)ga;                              // global_addr[31:0]
    g0[3] = (unsigned)((ga >> 32) & 0x1FFFFFFu) | (2u << 30);  // [56:32] | type=2
    i32x8 g1;
    g1[0] = (1 << 16) | (1 << 20) | (4 << 22) | (3 << 25); // dsz=2B,pad,int,amt
    g1[1] = (int)(64u << 16);                          // tensor_dim0 = 64
    g1[2] = (int)(512u << 16);                         // tensor_dim1 = 512
    g1[3] = (int)(64u << 16);                          // tile_dim0 = 64
    g1[4] = 512;                                       // tile_dim1 = 512
    g1[5] = 64;                                        // tensor_dim0_stride = 64
    g1[6] = 0;
    g1[7] = 0;
    i32x4 z4 = {0, 0, 0, 0};
    i32x8 z8 = {0, 0, 0, 0, 0, 0, 0, 0};
    __builtin_amdgcn_tensor_load_to_lds(g0, g1, z4, z4, z8, 0);
    __builtin_amdgcn_s_wait_tensorcnt(0);
  }
  __syncthreads();

  const bf16* qbase = qh + ((size_t)((b * HEADS + h) * T_TXT + i0) << 6);
  const v16bf aq0 = load_frag_a(qbase, DHEAD, lane);        // d = 0..31
  const v16bf aq1 = load_frag_a(qbase + 32, DHEAD, lane);   // d = 32..63

  // per-row key windows [lo, hi): mask == (cm >= mt) & (im < mt), mt = j/64 + 1
  int   lo[8], hi[8];
  bool  emp[8];
  float mrow[8], lrow[8];
#pragma unroll
  for (int r = 0; r < 8; ++r) {
    const int i  = i0 + r + 8 * half;
    int imx = imv[b * T_TXT + i];
    int cmx = cmv[b * T_TXT + i];
    lo[r] = (imx > 9 ? 9 : imx) * N_IMG;
    hi[r] = (cmx > 8 ? 8 : cmx) * N_IMG;
    emp[r] = (lo[r] >= hi[r]);          // fully masked row -> uniform softmax
    if (emp[r]) { lo[r] = 0; hi[r] = T_KEY; }
    mrow[r] = NEG_BIG; lrow[r] = 0.f;
  }
  // wave-uniform union of windows -> skip dead key chunks
  int tlo = T_KEY, thi = 0;
#pragma unroll
  for (int r = 0; r < 8; ++r) { tlo = min(tlo, lo[r]); thi = max(thi, hi[r]); }
#pragma unroll
  for (int o = 1; o < 32; o <<= 1) {
    tlo = min(tlo, __shfl_xor(tlo, o, 32));
    thi = max(thi, __shfl_xor(thi, o, 32));
  }

  v8f oacc[4] = {};

  for (int jc = 0; jc < 8; ++jc) {
    const int j0 = jc * 64;
    if (j0 >= thi || j0 + 64 <= tlo) continue;   // wave-uniform branch

    // ---- S = q @ k^T for 16 rows x 64 keys (8 wmma, K from padded LDS) ----
    v8f s[4] = {};
#pragma unroll
    for (int c = 0; c < 4; ++c) {
      const bf16* krow = Ksh + (size_t)(j0 + 16 * c) * KPAD;
      v16bf bk0 = load_frag_b(krow, KPAD, lane);
      v16bf bk1 = load_frag_b(krow + 32, KPAD, lane);
      s[c] = wmma_bf16(aq0, bk0, s[c]);
      s[c] = wmma_bf16(aq1, bk1, s[c]);
    }

    // ---- mask + online softmax ----
    float cmax[8];
#pragma unroll
    for (int r = 0; r < 8; ++r) cmax[r] = NEG_BIG;
#pragma unroll
    for (int c = 0; c < 4; ++c)
#pragma unroll
      for (int r = 0; r < 8; ++r) {
        const int j = j0 + 16 * c + nl;
        float v = s[c][r];
        v = emp[r] ? 0.f : ((j >= lo[r] && j < hi[r]) ? v : NEG_BIG);
        s[c][r] = v;
        cmax[r] = fmaxf(cmax[r], v);
      }
#pragma unroll
    for (int o = 1; o < 16; o <<= 1)
#pragma unroll
      for (int r = 0; r < 8; ++r)
        cmax[r] = fmaxf(cmax[r], __shfl_xor(cmax[r], o, 32));

    float alpha[8];
#pragma unroll
    for (int r = 0; r < 8; ++r) {
      const float mn = fmaxf(mrow[r], cmax[r]);
      alpha[r] = __expf(mrow[r] - mn);
      mrow[r]  = mn;
    }
    float ls[8];
#pragma unroll
    for (int r = 0; r < 8; ++r) ls[r] = 0.f;
#pragma unroll
    for (int c = 0; c < 4; ++c)
#pragma unroll
      for (int r = 0; r < 8; ++r) {
        const float p = __expf(s[c][r] - mrow[r]);
        ls[r] += p;
        P[r + 8 * half][16 * c + nl] = (bf16)p;   // C-layout -> LDS row-major
      }
#pragma unroll
    for (int o = 1; o < 16; o <<= 1)
#pragma unroll
      for (int r = 0; r < 8; ++r) ls[r] += __shfl_xor(ls[r], o, 32);
#pragma unroll
    for (int r = 0; r < 8; ++r) lrow[r] = lrow[r] * alpha[r] + ls[r];
#pragma unroll
    for (int dt = 0; dt < 4; ++dt)
#pragma unroll
      for (int r = 0; r < 8; ++r) oacc[dt][r] *= alpha[r];

    // cross-lane LDS dependency: all P stores of this wave must land
    asm volatile("s_wait_dscnt 0" ::: "memory");

    // ---- O += P @ V (8 wmma): A from LDS, B from vT[b][h][d][j] ----
#pragma unroll
    for (int ks = 0; ks < 2; ++ks) {
      v16bf apf = load_frag_a(&P[0][0] + ks * 32, KPAD, lane);
#pragma unroll
      for (int dt = 0; dt < 4; ++dt) {
        v16bf bv = load_frag_b(vbase + ((size_t)(dt * 16) * T_KEY) + j0 + ks * 32,
                               T_KEY, lane);
        oacc[dt] = wmma_bf16(apf, bv, oacc[dt]);
      }
    }
  }

  // ---- normalize and write O[b][i][h*64+d] bf16 ----
#pragma unroll
  for (int dt = 0; dt < 4; ++dt)
#pragma unroll
    for (int r = 0; r < 8; ++r) {
      const int i = i0 + r + 8 * half;
      const int d = dt * 16 + nl;
      ob[((size_t)(b * T_TXT + i) << 10) + h * DHEAD + d] =
          (bf16)(oacc[dt][r] / lrow[r]);
    }
}

// ---------------------------------------------------------------------------
// Host launcher
// ---------------------------------------------------------------------------
extern "C" void kernel_launch(void* const* d_in, const int* in_sizes, int n_in,
                              void* d_out, int out_size, void* d_ws, size_t ws_size,
                              hipStream_t stream) {
  (void)in_sizes; (void)n_in; (void)out_size; (void)ws_size;
  const float*         x     = (const float*)d_in[0];
  const float*         media = (const float*)d_in[1];
  const unsigned char* ml    = (const unsigned char*)d_in[2];
  const unsigned char* cl    = (const unsigned char*)d_in[3];
  const float*         g     = (const float*)d_in[4];
  const float*         be    = (const float*)d_in[5];
  const float*         Wq    = (const float*)d_in[6];
  const float*         Wkv   = (const float*)d_in[7];
  const float*         Wo    = (const float*)d_in[8];
  float*               out   = (float*)d_out;

  char* ws = (char*)d_ws;
  size_t off = 0;
  auto alloc = [&](size_t bytes) {
    void* p = ws + off;
    off += (bytes + 255) & ~(size_t)255;
    return p;
  };
  const size_t M_TXT = (size_t)B_SZ * T_TXT;          // 8192
  const size_t M_KEY = (size_t)B_SZ * T_KEY;          // 2048
  bf16* XN   = (bf16*)alloc(M_TXT * D_MODEL * 2);     // 64 MB
  bf16* MB   = (bf16*)alloc(M_KEY * DV * 2);          // 4 MB
  bf16* WQT  = (bf16*)alloc((size_t)INNER * D_MODEL * 2);     // 8 MB (N=1024,K=4096)
  bf16* WKVT = (bf16*)alloc((size_t)2 * INNER * DV * 2);      // 4 MB (N=2048,K=1024)
  bf16* WOT  = (bf16*)alloc((size_t)D_MODEL * INNER * 2);     // 8 MB (N=4096,K=1024)
  bf16* QH   = (bf16*)alloc(M_TXT * INNER * 2);       // 16 MB
  bf16* KH   = (bf16*)alloc(M_KEY * INNER * 2);       // 4 MB
  bf16* VT   = (bf16*)alloc(M_KEY * INNER * 2);       // 4 MB
  bf16* OB   = (bf16*)alloc(M_TXT * INNER * 2);       // 16 MB
  int*  CM   = (int*)alloc(M_TXT * 4);
  int*  IM   = (int*)alloc(M_TXT * 4);

  // Stage 1: precision/layout prep + masks
  ln_kernel<<<(int)M_TXT, 256, 0, stream>>>(x, g, be, XN);
  {
    int n = (int)(M_KEY * DV);
    convert_bf16<<<(n + 255) / 256, 256, 0, stream>>>(media, MB, n);
  }
  transpose_bf16<<<(D_MODEL * INNER + 255) / 256, 256, 0, stream>>>(Wq,  WQT, D_MODEL, INNER);
  transpose_bf16<<<(DV * 2 * INNER + 255) / 256, 256, 0, stream>>>(Wkv, WKVT, DV, 2 * INNER);
  transpose_bf16<<<(INNER * D_MODEL + 255) / 256, 256, 0, stream>>>(Wo,  WOT, INNER, D_MODEL);
  mask_kernel<<<1, B_SZ, 0, stream>>>(ml, cl, CM, IM);

  // Stage 2: projections (WMMA GEMMs, 64x64 wave tiles)
  gemm_bf16_kernel<EpiQ><<<dim3(INNER / 128, (int)(M_TXT / 256)), 256, 0, stream>>>(
      XN, WQT, D_MODEL, EpiQ{QH});
  gemm_bf16_kernel<EpiKV><<<dim3(2 * INNER / 128, (int)(M_KEY / 256)), 256, 0, stream>>>(
      MB, WKVT, DV, EpiKV{KH, VT});

  // Stage 3: masked flash attention (TDM-staged K, WMMA QK^T and PV)
  attn_kernel<<<(B_SZ * HEADS * (T_TXT / 16)) / 8, 256, 0, stream>>>(QH, KH, VT, CM, IM, OB);

  // Stage 4: output projection (WMMA GEMM, f32 out)
  gemm_bf16_kernel<EpiO><<<dim3(D_MODEL / 128, (int)(M_TXT / 256)), 256, 0, stream>>>(
      OB, WOT, INNER, EpiO{out});
}
